// BiMambaSmooth_39359080301037
// MI455X (gfx1250) — matrix operand order repaired and checked by reading, not compiled, not run on hardware
//
#include <hip/hip_runtime.h>
#include <hip/hip_bf16.h>
#include <math.h>

// Problem dims (fixed by the reference)
#define BDIM   2
#define LSEQ   2048
#define DM     256
#define NLAYERS 4
#define DI     512
#define NS     16
#define DTR    16
#define ROWS   (BDIM * LSEQ)   // 4096 token rows

typedef __attribute__((ext_vector_type(16))) __bf16 v16bf;
typedef __attribute__((ext_vector_type(8)))  float  v8f;
typedef __attribute__((ext_vector_type(4)))  unsigned int v4u;
typedef __attribute__((ext_vector_type(8)))  int    v8i_;
typedef __attribute__((ext_vector_type(4)))  int    v4i_;

__device__ __forceinline__ float siluf(float x)     { return x / (1.0f + __expf(-x)); }
__device__ __forceinline__ float softplusf(float x) { return (x > 20.0f) ? x : log1pf(__expf(x)); }
__device__ __forceinline__ float geluf(float x)     { return 0.5f * x * (1.0f + erff(x * 0.70710678118654752f)); }

// ---------------------------------------------------------------------------
// WMMA GEMM, compile-time shape:  C[ROWS,N] = act( A[:,a_off:a_off+K] @ W[N,K]^T + bias )
// Block = 256 threads = 8 wave32s; block tile 32 rows x 128 cols.
// Wave w: row-half = w>>2, col-quarter = w&3; each wave computes a 16x32 strip
// = 2x v_wmma_f32_16x16x32_bf16 sharing one A fragment per 32-wide K chunk.
// Templating K/N/ACT removes pad and bound guards where statically provable.
// ACT: 0 = none, 1 = softplus, 2 = gelu (exact)
// ---------------------------------------------------------------------------
template <int K, int N, int ACT>
__global__ __launch_bounds__(256) void gemm_wmma(
    const float* __restrict__ A, int lda, int a_off,
    const float* __restrict__ W, const float* __restrict__ bias,
    float* __restrict__ C, int ldc)
{
    __shared__ __bf16 sA[32 * 32];    // 2 KB
    __shared__ __bf16 sB[128 * 32];   // 8 KB

    const int tid   = threadIdx.x;
    const int lane  = tid & 31;
    const int wave  = tid >> 5;
    const int rHalf = wave >> 2;      // 0..1
    const int cQuad = wave & 3;       // 0..3
    const int m0    = blockIdx.x * 32;
    const int nBase = blockIdx.y * 128;

    v8f acc0 = {}, acc1 = {};

    for (int k0 = 0; k0 < K; k0 += 32) {
        if (k0 + 32 < K)  // pull next weight chunk toward L2 (global_prefetch_b8)
            __builtin_prefetch(&W[(size_t)nBase * K + k0 + 32], 0, 1);

        // Stage A: 32 rows x 32 k (fp32 -> bf16); pad only if K%32 != 0
        for (int e = tid; e < 32 * 32; e += 256) {
            int r = e >> 5, k = e & 31;
            float v;
            if (K % 32 == 0) v = A[(size_t)(m0 + r) * lda + a_off + k0 + k];
            else             v = (k0 + k < K) ? A[(size_t)(m0 + r) * lda + a_off + k0 + k] : 0.0f;
            sA[e] = (__bf16)v;
        }
        // Stage W: 128 n x 32 k; n-guard only if N%128 != 0
        for (int e = tid; e < 128 * 32; e += 256) {
            int r = e >> 5, k = e & 31;
            int n = nBase + r;
            float v;
            if (N % 128 == 0) v = W[(size_t)n * K + k0 + k];
            else              v = (n < N && (K % 32 == 0 || k0 + k < K)) ? W[(size_t)n * K + k0 + k] : 0.0f;
            sB[e] = (__bf16)v;
        }
        __syncthreads();

        // A fragment: lane = M row; lane-half selects K 0-15 / 16-31
        v16bf a, b0, b1;
        const int aRow   = rHalf * 16 + (lane & 15);
        const int kaBase = (lane >> 4) << 4;
        #pragma unroll
        for (int i = 0; i < 16; ++i) {
            int ka = kaBase + i;
            a[i] = sA[aRow * 32 + ka];
        }
        // B fragments: lane = N col; lane-half selects K 0-15 / 16-31
        const int nc0 = cQuad * 32 + (lane & 15);
        const int kb  = (lane >> 4) << 4;
        #pragma unroll
        for (int i = 0; i < 16; ++i) {
            b0[i] = sB[nc0 * 32 + kb + i];
            b1[i] = sB[(nc0 + 16) * 32 + kb + i];
        }
        acc0 = __builtin_amdgcn_wmma_f32_16x16x32_bf16(false, a, false, b0, (short)0, acc0, false, false);
        acc1 = __builtin_amdgcn_wmma_f32_16x16x32_bf16(false, a, false, b1, (short)0, acc1, false, false);
        __syncthreads();
    }

    // Store: C layout lane = N col, VGPR j = M row (lane halves = M 0-7 / 8-15)
    #pragma unroll
    for (int t = 0; t < 2; ++t) {
        const int n = nBase + cQuad * 32 + t * 16 + (lane & 15);
        if (N % 128 == 0 || n < N) {
            float bv = bias ? bias[n] : 0.0f;
            #pragma unroll
            for (int j = 0; j < 8; ++j) {
                int m = m0 + rHalf * 16 + ((lane >> 4) << 3) + j;
                float v = (t ? acc1[j] : acc0[j]) + bv;
                if (ACT == 1) v = softplusf(v);
                if (ACT == 2) v = geluf(v);
                C[(size_t)m * ldc + n] = v;
            }
        }
    }
}

// ---------------------------------------------------------------------------
// Input projection: (ROWS,4) @ (256,4)^T + b -> (ROWS,256). K=4, VALU is fine.
// ---------------------------------------------------------------------------
__global__ void inproj_kernel(const float* __restrict__ in,
                              const float* __restrict__ pw,
                              const float* __restrict__ pb,
                              float* __restrict__ out)
{
    int idx = blockIdx.x * blockDim.x + threadIdx.x;   // ROWS*DM
    if (idx >= ROWS * DM) return;
    int c = idx & (DM - 1);
    int m = idx >> 8;
    float acc = pb[c];
    #pragma unroll
    for (int k = 0; k < 4; ++k) acc += in[m * 4 + k] * pw[c * 4 + k];
    out[idx] = acc;
}

// ---------------------------------------------------------------------------
// LayerNorm over DM=256 (one block per row, biased variance like jnp.var)
// ---------------------------------------------------------------------------
__global__ __launch_bounds__(256) void layernorm_kernel(
    const float* __restrict__ x, const float* __restrict__ w,
    const float* __restrict__ bvec, float* __restrict__ out)
{
    __shared__ float r1[256], r2[256];
    int row = blockIdx.x;
    int c = threadIdx.x;
    float v = x[(size_t)row * DM + c];
    r1[c] = v; r2[c] = v * v;
    __syncthreads();
    for (int s = 128; s > 0; s >>= 1) {
        if (c < s) { r1[c] += r1[c + s]; r2[c] += r2[c + s]; }
        __syncthreads();
    }
    float m   = r1[0] * (1.0f / DM);
    float var = r2[0] * (1.0f / DM) - m * m;
    float inv = rsqrtf(var + 1e-5f);
    out[(size_t)row * DM + c] = (v - m) * inv * w[c] + bvec[c];
}

// ---------------------------------------------------------------------------
// Causal depthwise conv1d (K=4, correlation, left pad 3) + SiLU.
// ---------------------------------------------------------------------------
__global__ void conv_silu_kernel(const float* __restrict__ xz,
                                 const float* __restrict__ cw,
                                 const float* __restrict__ cb,
                                 float* __restrict__ xcs)
{
    int idx = blockIdx.x * blockDim.x + threadIdx.x;   // ROWS*DI, d fastest
    if (idx >= ROWS * DI) return;
    int d  = idx & (DI - 1);
    int bl = idx >> 9;
    int b = bl / LSEQ, l = bl % LSEQ;
    float acc = cb[d];
    #pragma unroll
    for (int j = 0; j < 4; ++j) {
        int ls = l - 3 + j;
        if (ls >= 0) acc += cw[d * 4 + j] * xz[(size_t)(b * LSEQ + ls) * 1024 + d];
    }
    xcs[idx] = siluf(acc);
}

// ---------------------------------------------------------------------------
// Selective scan. One thread = one (b,d) channel, 16 fp32 states in VGPRs.
// The per-chunk B/C tile (64 steps x 32 f32, row stride 48) is fetched by the
// Tensor Data Mover: wave 0 issues tensor_load_to_lds with a 2-D D#
// (ISA 8.3/8.4: count=1, type=2, data_size=4B, tensor 32x64 stride 48,
// tile 32x64), waits s_wait_tensorcnt 0, block barriers, all waves consume.
// ---------------------------------------------------------------------------
__global__ __launch_bounds__(64) void scan_kernel(
    const float* __restrict__ dt,    // (ROWS, DI) post-softplus
    const float* __restrict__ dbc,   // (ROWS, 48): [dt_lowrank | B | C]
    const float* __restrict__ xcs,   // (ROWS, DI)
    const float* __restrict__ xz,    // (ROWS, 1024), z = cols 512..1023
    const float* __restrict__ Alog,  // (DI, NS) for this layer/dir
    const float* __restrict__ Dp,    // (DI)
    float* __restrict__ y)           // (ROWS, DI)
{
    __shared__ float sBC[64 * 32];   // [step][0..15 = B, 16..31 = C]
    const int b = blockIdx.y;
    const int d = blockIdx.x * 64 + threadIdx.x;

    float Acoef[NS], h[NS];
    #pragma unroll
    for (int n = 0; n < NS; ++n) { Acoef[n] = -__expf(Alog[d * NS + n]); h[n] = 0.0f; }
    const float dskip = Dp[d];
    const unsigned ldsOff = (unsigned)(unsigned long long)(uintptr_t)(void*)sBC;

    for (int l0 = 0; l0 < LSEQ; l0 += 64) {
        if ((threadIdx.x >> 5) == 0) {   // wave 0 drives the TDM
            unsigned long long ga =
                (unsigned long long)(uintptr_t)&dbc[(size_t)(b * LSEQ + l0) * 48 + 16];
            v4u g0;
            g0[0] = 1u;                                   // count=1 (valid D#)
            g0[1] = ldsOff;                               // lds_addr
            g0[2] = (unsigned)(ga & 0xFFFFFFFFu);         // global_addr lo
            g0[3] = (unsigned)((ga >> 32) & 0x1FFFFFFu)   // global_addr hi (57b)
                    | (2u << 30);                         // type = 2 ("image")
            v8i_ g1;
            g1[0] = 0x00020000;        // workgroup_mask=0, data_size=2 (4 bytes)
            g1[1] = (32 << 16);        // tensor_dim0 = 32 (lo16)
            g1[2] = (64 << 16);        // tensor_dim0 hi=0 | tensor_dim1 = 64
            g1[3] = (32 << 16);        // tensor_dim1 hi=0 | tile_dim0 = 32
            g1[4] = 64;                // tile_dim1 = 64, tile_dim2 = 0
            g1[5] = 48;                // tensor_dim0_stride = 48 (row pitch)
            g1[6] = 0;                 // stride hi / tensor_dim1_stride lo
            g1[7] = 0;
            v4i_ gz  = {0, 0, 0, 0};             // groups 2/3 unused (2-D tensor)
            v8i_ gz8 = {0, 0, 0, 0, 0, 0, 0, 0}; // extra group (clang-23 6-arg form)
            __builtin_amdgcn_tensor_load_to_lds(g0, g1, gz, gz, gz8, 0);
            __builtin_amdgcn_s_wait_tensorcnt(0);
        }
        __syncthreads();
        for (int s = 0; s < 64; ++s) {
            size_t row = (size_t)(b * LSEQ + l0 + s);
            float dtv = dt[row * DI + d];
            float xv  = xcs[row * DI + d];
            float dbx = dtv * xv;
            float accy = 0.0f;
            #pragma unroll
            for (int n = 0; n < NS; ++n) {
                float dA = __expf(dtv * Acoef[n]);
                h[n] = dA * h[n] + dbx * sBC[s * 32 + n];
                accy += h[n] * sBC[s * 32 + 16 + n];
            }
            float z = xz[row * 1024 + 512 + d];
            y[row * DI + d] = (accy + dskip * xv) * siluf(z);
        }
        __syncthreads();
    }
}

// ---------------------------------------------------------------------------
// Sequence reversal helpers (direction-agnostic pipeline reuse)
// ---------------------------------------------------------------------------
__global__ void reverse_kernel(const float* __restrict__ src, float* __restrict__ dst)
{
    int idx = blockIdx.x * blockDim.x + threadIdx.x;   // ROWS*DM
    if (idx >= ROWS * DM) return;
    int c  = idx & (DM - 1);
    int bl = idx >> 8;
    int b = bl / LSEQ, l = bl % LSEQ;
    dst[idx] = src[(size_t)(b * LSEQ + (LSEQ - 1 - l)) * DM + c];
}

__global__ void add_rev_kernel(const float* __restrict__ f,
                               const float* __restrict__ r,
                               float* __restrict__ h)
{
    int idx = blockIdx.x * blockDim.x + threadIdx.x;   // ROWS*DM
    if (idx >= ROWS * DM) return;
    int c  = idx & (DM - 1);
    int bl = idx >> 8;
    int b = bl / LSEQ, l = bl % LSEQ;
    h[idx] = f[idx] + r[(size_t)(b * LSEQ + (LSEQ - 1 - l)) * DM + c];
}

// ---------------------------------------------------------------------------
// Orchestration
// ---------------------------------------------------------------------------
extern "C" void kernel_launch(void* const* d_in, const int* in_sizes, int n_in,
                              void* d_out, int out_size, void* d_ws, size_t ws_size,
                              hipStream_t stream)
{
    (void)in_sizes; (void)n_in; (void)ws_size;

    const float* in_feat    = (const float*)d_in[0];
    const float* proj_w     = (const float*)d_in[1];
    const float* proj_b     = (const float*)d_in[2];
    const float* in_proj_w  = (const float*)d_in[3];   // (NL, 1024, 256)
    const float* in_proj_b  = (const float*)d_in[4];   // (NL, 1024)
    const float* out_proj_w = (const float*)d_in[5];   // (NL, 256, 512)
    const float* out_proj_b = (const float*)d_in[6];   // (NL, 256)
    const float* conv_w     = (const float*)d_in[7];   // (NL, 2, 512, 4)
    const float* conv_b     = (const float*)d_in[8];   // (NL, 2, 512)
    const float* x_proj_w   = (const float*)d_in[9];   // (NL, 2, 48, 512)
    const float* dt_proj_w  = (const float*)d_in[10];  // (NL, 2, 512, 16)
    const float* dt_proj_b  = (const float*)d_in[11];  // (NL, 2, 512)
    const float* A_log      = (const float*)d_in[12];  // (NL, 2, 512, 16)
    const float* D_skip     = (const float*)d_in[13];  // (NL, 2, 512)
    const float* ln_w       = (const float*)d_in[14];  // (NL, 256)
    const float* ln_b       = (const float*)d_in[15];
    const float* norm_f_w   = (const float*)d_in[16];
    const float* norm_f_b   = (const float*)d_in[17];
    const float* h1_w       = (const float*)d_in[18];  // (512, 256)
    const float* h1_b       = (const float*)d_in[19];
    const float* h2_w       = (const float*)d_in[20];  // (4, 512)
    const float* h2_b       = (const float*)d_in[21];

    float* ws = (float*)d_ws;
    size_t off = 0;
    float* h    = ws + off; off += (size_t)ROWS * DM;
    float* hn   = ws + off; off += (size_t)ROWS * DM;
    float* hnr  = ws + off; off += (size_t)ROWS * DM;
    float* xz   = ws + off; off += (size_t)ROWS * 1024;
    float* xcs  = ws + off; off += (size_t)ROWS * DI;
    float* dbc  = ws + off; off += (size_t)ROWS * 48;
    float* dtb  = ws + off; off += (size_t)ROWS * DI;
    float* yb   = ws + off; off += (size_t)ROWS * DI;
    float* outf = ws + off; off += (size_t)ROWS * DM;
    float* outr = ws + off; off += (size_t)ROWS * DM;
    float* h1o  = ws + off; off += (size_t)ROWS * 512;

    const int NEL = ROWS * DM;
    inproj_kernel<<<(NEL + 255) / 256, 256, 0, stream>>>(in_feat, proj_w, proj_b, h);

    for (int i = 0; i < NLAYERS; ++i) {
        layernorm_kernel<<<ROWS, 256, 0, stream>>>(h, ln_w + i * DM, ln_b + i * DM, hn);
        reverse_kernel<<<(NEL + 255) / 256, 256, 0, stream>>>(hn, hnr);

        for (int dir = 0; dir < 2; ++dir) {
            const float* src = dir ? hnr : hn;
            const int pd = i * 2 + dir;

            // in_proj: (ROWS,256) -> (ROWS,1024)
            gemm_wmma<DM, 1024, 0><<<dim3(ROWS / 32, 1024 / 128), 256, 0, stream>>>(
                src, DM, 0, in_proj_w + (size_t)i * 1024 * DM, in_proj_b + i * 1024,
                xz, 1024);

            conv_silu_kernel<<<(ROWS * DI + 255) / 256, 256, 0, stream>>>(
                xz, conv_w + (size_t)pd * DI * 4, conv_b + pd * DI, xcs);

            // x_proj: (ROWS,512) -> (ROWS,48)
            gemm_wmma<DI, 48, 0><<<dim3(ROWS / 32, 1), 256, 0, stream>>>(
                xcs, DI, 0, x_proj_w + (size_t)pd * 48 * DI, nullptr, dbc, 48);

            // dt_proj (+softplus): cols 0..15 of dbc -> (ROWS,512)
            gemm_wmma<DTR, DI, 1><<<dim3(ROWS / 32, DI / 128), 256, 0, stream>>>(
                dbc, 48, 0, dt_proj_w + (size_t)pd * DI * DTR, dt_proj_b + pd * DI,
                dtb, DI);

            scan_kernel<<<dim3(DI / 64, BDIM), 64, 0, stream>>>(
                dtb, dbc, xcs, xz, A_log + (size_t)pd * DI * NS, D_skip + pd * DI, yb);

            // out_proj: (ROWS,512) -> (ROWS,256)
            gemm_wmma<DI, DM, 0><<<dim3(ROWS / 32, DM / 128), 256, 0, stream>>>(
                yb, DI, 0, out_proj_w + (size_t)i * DM * DI, out_proj_b + i * DM,
                dir ? outr : outf, DM);
        }
        add_rev_kernel<<<(NEL + 255) / 256, 256, 0, stream>>>(outf, outr, h);
    }

    // Final norm + MLP head
    layernorm_kernel<<<ROWS, 256, 0, stream>>>(h, norm_f_w, norm_f_b, hn);
    gemm_wmma<DM, 512, 2><<<dim3(ROWS / 32, 512 / 128), 256, 0, stream>>>(
        hn, DM, 0, h1_w, h1_b, h1o, 512);                      // GELU fused
    gemm_wmma<512, 4, 0><<<dim3(ROWS / 32, 1), 256, 0, stream>>>(
        h1o, 512, 0, h2_w, h2_b, (float*)d_out, 4);

    // Mask output (all zeros) occupies the tail of d_out, if present
    if (out_size > ROWS * 4) {
        (void)hipMemsetAsync((char*)d_out + (size_t)ROWS * 4 * sizeof(float), 0,
                             (size_t)(out_size - ROWS * 4) * sizeof(float), stream);
    }
}
